// SelfAttention_86182813762323
// MI455X (gfx1250) — compile-verified, matrix operand order checked by reading, not run on hardware
//
#include <hip/hip_runtime.h>
#include <hip/hip_bf16.h>
#include <stdint.h>

// Problem constants (B=4, S=2048, H=1024 from the reference)
#define BATCH 4
#define SEQ   2048
#define HID   1024
#define N3H   3072

typedef __attribute__((ext_vector_type(16))) _Float16 v16h;
typedef __attribute__((ext_vector_type(8)))  float    v8f;
typedef __attribute__((ext_vector_type(8)))  _Float16 h8;
typedef __attribute__((ext_vector_type(4)))  float    f4;

// ---------------------------------------------------------------------------
// WMMA 16x16x32 f16 fragment layout (CDNA5 ISA 7.12.2, wave32):
//   lane L owns row/col (L&15); K elements = {kb..kb+7} and {kb+16..kb+23},
//   kb = (L<16) ? 0 : 8.  kpos passed below already includes k0 + kb.
// Loads 4 adjacent 16-row fragments (a 64-row slab) in one call so the
// compiler can cluster the 8 b128 loads and hide latency.
// ---------------------------------------------------------------------------
static __device__ inline void load_frag4(v16h f[4], const _Float16* __restrict__ base,
                                         int stride, int row0, int idx, int kpos) {
#pragma unroll
  for (int i = 0; i < 4; ++i) {
    const _Float16* p = base + (size_t)(row0 + i * 16 + idx) * stride + kpos;
    h8 lo = *(const h8*)(p);
    h8 hi = *(const h8*)(p + 16);
#pragma unroll
    for (int j = 0; j < 8; ++j) { f[i][j] = lo[j]; f[i][j + 8] = hi[j]; }
  }
}

static __device__ inline void mma4x4(v8f acc[4][4], const v16h a[4], const v16h b[4]) {
#pragma unroll
  for (int mi = 0; mi < 4; ++mi)
#pragma unroll
    for (int ni = 0; ni < 4; ++ni)
      acc[mi][ni] = __builtin_amdgcn_wmma_f32_16x16x32_f16(
          false, a[mi], false, b[ni], (short)0, acc[mi][ni], false, false);
}

// ---------------------------------------------------------------------------
// Kernel 0: X f32 -> Xh f16 (one-shot downconvert; removes all cvt from the
// GEMM hot loops and halves A-side read traffic).
// ---------------------------------------------------------------------------
__global__ __launch_bounds__(256)
void convert_x(const float* __restrict__ X, _Float16* __restrict__ Xh) {
  const size_t i = ((size_t)blockIdx.x * 256 + threadIdx.x) * 8;
  f4 a = *(const f4*)(X + i);
  f4 b = *(const f4*)(X + i + 4);
  h8 o;
#pragma unroll
  for (int j = 0; j < 4; ++j) { o[j] = (_Float16)a[j]; o[j + 4] = (_Float16)b[j]; }
  *(h8*)(Xh + i) = o;
}

// ---------------------------------------------------------------------------
// Kernel 1: W (1024 x 3072 f32) -> Wt (3072 x 1024 f16) via LDS transpose,
// so GEMM B-fragments (one column per lane) become row-contiguous loads.
// ---------------------------------------------------------------------------
__global__ __launch_bounds__(256)
void transpose_w(const float* __restrict__ W, _Float16* __restrict__ Wt) {
  __shared__ float tile[32][33];
  const int n0 = blockIdx.x * 32;
  const int k0 = blockIdx.y * 32;
#pragma unroll
  for (int i = 0; i < 4; ++i) {
    int k = threadIdx.y + i * 8;
    tile[k][threadIdx.x] = W[(size_t)(k0 + k) * N3H + n0 + threadIdx.x];
  }
  __syncthreads();
#pragma unroll
  for (int i = 0; i < 4; ++i) {
    int n = threadIdx.y + i * 8;
    Wt[(size_t)(n0 + n) * HID + k0 + threadIdx.x] =
        (_Float16)tile[threadIdx.x][n];
  }
}

// ---------------------------------------------------------------------------
// Shared GEMM core: 4 waves/block in a 2x2 grid; each wave owns a 64x64 tile
// (4x4 WMMA f32 accumulators).  K-loop is double-buffered: fragments for
// K-tile k+1 are in flight while the 16 WMMAs for tile k issue.
// ---------------------------------------------------------------------------
#define GEMM_PIPELINE(Abase, Astride, arow0, Bbase, Bstride, brow0, KDIM)       \
  v8f acc[4][4] = {};                                                           \
  {                                                                             \
    v16h a0[4], b0[4], a1[4], b1[4];                                            \
    load_frag4(a0, (Abase), (Astride), (arow0), idx, kb);                       \
    load_frag4(b0, (Bbase), (Bstride), (brow0), idx, kb);                       \
    for (int k0 = 0; k0 < (KDIM); k0 += 64) {                                   \
      if (k0 + 128 < (KDIM)) {                                                  \
        __builtin_prefetch((Abase) + (size_t)((arow0) + idx) * (Astride) +      \
                               k0 + 128, 0, 1);                                 \
        __builtin_prefetch((Bbase) + (size_t)((brow0) + idx) * (Bstride) +      \
                               k0 + 128, 0, 1);                                 \
      }                                                                         \
      load_frag4(a1, (Abase), (Astride), (arow0), idx, k0 + 32 + kb);           \
      load_frag4(b1, (Bbase), (Bstride), (brow0), idx, k0 + 32 + kb);           \
      mma4x4(acc, a0, b0);                                                      \
      if (k0 + 64 < (KDIM)) {                                                   \
        load_frag4(a0, (Abase), (Astride), (arow0), idx, k0 + 64 + kb);         \
        load_frag4(b0, (Bbase), (Bstride), (brow0), idx, k0 + 64 + kb);         \
      }                                                                         \
      mma4x4(acc, a1, b1);                                                      \
    }                                                                           \
  }

// ---------------------------------------------------------------------------
// Kernel 2: QKV = Xh @ Wt^T + b.  Q stored pre-scaled by 1/sqrt(H);
// V stored transposed (Vt[d][s]) so the P.V GEMM gets contiguous B rows.
// ---------------------------------------------------------------------------
__global__ __launch_bounds__(128)
void qkv_gemm(const _Float16* __restrict__ Xh, const _Float16* __restrict__ Wt,
              const float* __restrict__ bias,
              _Float16* __restrict__ Qh, _Float16* __restrict__ Kh,
              _Float16* __restrict__ Vt) {
  const int lane = threadIdx.x & 31;
  const int wave = threadIdx.x >> 5;
  const int idx  = lane & 15;
  const int kb   = (lane >> 4) << 3;                 // 0 or 8
  const int m0   = blockIdx.y * 128 + (wave & 1) * 64;
  const int n0   = blockIdx.x * 128 + (wave >> 1) * 64;

  GEMM_PIPELINE(Xh, HID, m0, Wt, HID, n0, HID)

  const int mrow_add = (lane < 16) ? 0 : 8;          // C layout: VGPR r -> M=r | r+8
#pragma unroll
  for (int mi = 0; mi < 4; ++mi)
#pragma unroll
    for (int ni = 0; ni < 4; ++ni) {
      const int   ng = n0 + ni * 16 + idx;
      const float bv = bias[ng];
#pragma unroll
      for (int r = 0; r < 8; ++r) {
        const int mg = m0 + mi * 16 + r + mrow_add;
        const int bb = mg >> 11;                     // / SEQ
        const int s  = mg & (SEQ - 1);
        const float v = acc[mi][ni][r] + bv;
        if (ng < HID) {                              // Q, pre-scaled by 1/32
          Qh[((size_t)bb * SEQ + s) * HID + ng] = (_Float16)(v * 0.03125f);
        } else if (ng < 2 * HID) {                   // K
          Kh[((size_t)bb * SEQ + s) * HID + (ng - HID)] = (_Float16)v;
        } else {                                     // V, transposed
          Vt[((size_t)bb * HID + (ng - 2 * HID)) * SEQ + s] = (_Float16)v;
        }
      }
    }
}

// ---------------------------------------------------------------------------
// Kernel 3: scores[b,q,c] = Qh[b,q,:] . Kh[b,c,:]   (+ key mask -> -inf)
// ---------------------------------------------------------------------------
__global__ __launch_bounds__(128)
void scores_gemm(const _Float16* __restrict__ Qh, const _Float16* __restrict__ Kh,
                 const uint8_t* __restrict__ mask, float* __restrict__ Sc) {
  const int lane = threadIdx.x & 31;
  const int wave = threadIdx.x >> 5;
  const int idx  = lane & 15;
  const int kb   = (lane >> 4) << 3;
  const int b    = blockIdx.z;
  const int q0   = blockIdx.y * 128 + (wave & 1) * 64;
  const int c0   = blockIdx.x * 128 + (wave >> 1) * 64;

  const _Float16* Qb = Qh + (size_t)b * SEQ * HID;
  const _Float16* Kb = Kh + (size_t)b * SEQ * HID;

  GEMM_PIPELINE(Qb, HID, q0, Kb, HID, c0, HID)

  float* Sb = Sc + (size_t)b * SEQ * SEQ;
  const int mrow_add = (lane < 16) ? 0 : 8;
#pragma unroll
  for (int mi = 0; mi < 4; ++mi)
#pragma unroll
    for (int ni = 0; ni < 4; ++ni) {
      const int  cg = c0 + ni * 16 + idx;
      const bool mk = mask[(size_t)b * SEQ + cg] != 0;   // jnp.bool_: 1 byte
#pragma unroll
      for (int r = 0; r < 8; ++r) {
        const int qg = q0 + mi * 16 + r + mrow_add;
        Sb[(size_t)qg * SEQ + cg] = mk ? -__builtin_inff() : acc[mi][ni][r];
      }
    }
}

// ---------------------------------------------------------------------------
// Kernel 4: row softmax over k (2048 f32 per row), output f16 probabilities.
// ---------------------------------------------------------------------------
__global__ __launch_bounds__(256)
void softmax_rows(const float* __restrict__ Sc, _Float16* __restrict__ Ph) {
  const int row = blockIdx.x;                        // 0 .. BATCH*SEQ-1
  const float* src = Sc + (size_t)row * SEQ;
  _Float16*    dst = Ph + (size_t)row * SEQ;
  __shared__ float red[256];

  float v[8];
  float m = -__builtin_inff();
#pragma unroll
  for (int i = 0; i < 8; ++i) {
    v[i] = src[threadIdx.x + 256 * i];
    m = fmaxf(m, v[i]);
  }
  red[threadIdx.x] = m;
  __syncthreads();
  for (int off = 128; off > 0; off >>= 1) {
    if (threadIdx.x < off)
      red[threadIdx.x] = fmaxf(red[threadIdx.x], red[threadIdx.x + off]);
    __syncthreads();
  }
  m = red[0];
  __syncthreads();

  float sum = 0.f;
#pragma unroll
  for (int i = 0; i < 8; ++i) { v[i] = __expf(v[i] - m); sum += v[i]; }
  red[threadIdx.x] = sum;
  __syncthreads();
  for (int off = 128; off > 0; off >>= 1) {
    if (threadIdx.x < off) red[threadIdx.x] += red[threadIdx.x + off];
    __syncthreads();
  }
  const float inv = 1.0f / red[0];
#pragma unroll
  for (int i = 0; i < 8; ++i)
    dst[threadIdx.x + 256 * i] = (_Float16)(v[i] * inv);
}

// ---------------------------------------------------------------------------
// Kernel 5: Out[b,q,d] = sum_k P[b,q,k] * V[b,k,d]   (B-side reads Vt rows)
// ---------------------------------------------------------------------------
__global__ __launch_bounds__(128)
void pv_gemm(const _Float16* __restrict__ Ph, const _Float16* __restrict__ Vt,
             float* __restrict__ Out) {
  const int lane = threadIdx.x & 31;
  const int wave = threadIdx.x >> 5;
  const int idx  = lane & 15;
  const int kb   = (lane >> 4) << 3;
  const int b    = blockIdx.z;
  const int q0   = blockIdx.y * 128 + (wave & 1) * 64;
  const int d0   = blockIdx.x * 128 + (wave >> 1) * 64;

  const _Float16* Pb = Ph + (size_t)b * SEQ * SEQ;
  const _Float16* Vb = Vt + (size_t)b * HID * SEQ;

  GEMM_PIPELINE(Pb, SEQ, q0, Vb, SEQ, d0, SEQ)

  const int mrow_add = (lane < 16) ? 0 : 8;
#pragma unroll
  for (int mi = 0; mi < 4; ++mi)
#pragma unroll
    for (int ni = 0; ni < 4; ++ni) {
      const int dg = d0 + ni * 16 + idx;
#pragma unroll
      for (int r = 0; r < 8; ++r) {
        const int qg = q0 + mi * 16 + r + mrow_add;
        Out[((size_t)b * SEQ + qg) * HID + dg] = acc[mi][ni][r];
      }
    }
}

// ---------------------------------------------------------------------------
// Launcher.  Inputs: X (f32), attention_mask (bool/1B), W (f32), b (f32).
// Workspace (166 MB): Qh|Kh|Vt f16, scores f32, P f16, Wt f16, Xh f16.
// ---------------------------------------------------------------------------
extern "C" void kernel_launch(void* const* d_in, const int* in_sizes, int n_in,
                              void* d_out, int out_size, void* d_ws, size_t ws_size,
                              hipStream_t stream) {
  (void)in_sizes; (void)n_in; (void)out_size; (void)ws_size;
  const float*   X    = (const float*)d_in[0];
  const uint8_t* mask = (const uint8_t*)d_in[1];
  const float*   W    = (const float*)d_in[2];
  const float*   bias = (const float*)d_in[3];
  float*         Out  = (float*)d_out;

  char* ws = (char*)d_ws;
  size_t off = 0;
  _Float16* Qh = (_Float16*)(ws + off); off += (size_t)BATCH * SEQ * HID * 2;
  _Float16* Kh = (_Float16*)(ws + off); off += (size_t)BATCH * SEQ * HID * 2;
  _Float16* Vt = (_Float16*)(ws + off); off += (size_t)BATCH * HID * SEQ * 2;
  float*    Sc = (float*)   (ws + off); off += (size_t)BATCH * SEQ * SEQ * 4;
  _Float16* Ph = (_Float16*)(ws + off); off += (size_t)BATCH * SEQ * SEQ * 2;
  _Float16* Wt = (_Float16*)(ws + off); off += (size_t)N3H * HID * 2;
  _Float16* Xh = (_Float16*)(ws + off); off += (size_t)BATCH * SEQ * HID * 2;

  convert_x   <<<dim3((BATCH * SEQ * HID) / (256 * 8)), 256, 0, stream>>>(X, Xh);
  transpose_w <<<dim3(N3H / 32, HID / 32), dim3(32, 8), 0, stream>>>(W, Wt);
  qkv_gemm    <<<dim3(N3H / 128, (BATCH * SEQ) / 128), 128, 0, stream>>>(
      Xh, Wt, bias, Qh, Kh, Vt);
  scores_gemm <<<dim3(SEQ / 128, SEQ / 128, BATCH), 128, 0, stream>>>(
      Qh, Kh, mask, Sc);
  softmax_rows<<<dim3(BATCH * SEQ), 256, 0, stream>>>(Sc, Ph);
  pv_gemm     <<<dim3(HID / 128, SEQ / 128, BATCH), 128, 0, stream>>>(
      Ph, Vt, Out);
}
